// envelope_detector_3135326126326
// MI455X (gfx1250) — matrix operand order repaired and checked by reading, not compiled, not run on hardware
//
#include <hip/hip_runtime.h>
#include <stdint.h>

// ---------------- problem constants (from reference) ----------------
#define BATCH   32
#define CIN     64
#define T_IN    8192
#define CPC     5
#define OC      (CIN * CPC)      // 320
#define K1      50
#define K2      100
#define KK2     (CPC * K2)       // 500 reduction length -> permuted/padded to 512
#define T1      (T_IN - K1 + 1)  // 8143
#define T2      (T1 - K2 + 1)    // 8044

// ---------------- tiling ----------------
#define TILE_Z  512              // z outputs per workgroup (32 slices of 16)
#define YTIMES  640              // y samples computed per tile (TILE_Z + 99 -> 40 slices)
#define YROW    640              // ys row stride
#define XNEED   (TILE_Z + K2 - 1 + K1 - 1)  // 660 valid x samples
#define XS_SZ   704              // staged x entries (zero padded; max read 702)

typedef __attribute__((ext_vector_type(16))) __bf16 v16bf;
typedef __attribute__((ext_vector_type(8)))  float  v8f;

union ABReg { v16bf v; unsigned int u[8]; };

// f32 -> bf16 bits, round-to-nearest-even
static __device__ __forceinline__ unsigned short f2bf(float f) {
    union { float f; unsigned int u; } c; c.f = f;
    unsigned int u = c.u + 0x7FFFu + ((c.u >> 16) & 1u);
    return (unsigned short)(u >> 16);
}

// K-local offset within a 16-bit 16x32 A operand (ISA 7.12.2), per VGPR vr and lane-half h
static __device__ __forceinline__ int a_klocal(int vr, int h) {
    return (vr < 4) ? (8 * h + 2 * vr) : (16 + 8 * h + 2 * (vr - 4));
}

__global__ __launch_bounds__(256)
void envelope_fused_wmma(const float* __restrict__ x,
                         const float* __restrict__ w1,
                         const float* __restrict__ b1,
                         const float* __restrict__ w2,
                         const float* __restrict__ b2,
                         float* __restrict__ out) {
    // LDS staging (bf16 bit patterns in ushort); total ~27.4 KB
    __shared__ unsigned short xs[XS_SZ];          // x tile
    __shared__ unsigned short ys[6 * YROW];       // y rows 0..4, row 5 = zeros (K pad)
    __shared__ unsigned short w1s[16 * 64];       // stage-1 B: [n][k], zero pad n>=5, k>=50
    __shared__ unsigned short w2s[16 * 512];      // stage-2 B: [n][j], K-permuted (see below)

    const int tile = blockIdx.x;
    const int g    = blockIdx.y;
    const int b    = blockIdx.z;
    const int t0   = tile * TILE_Z;

    const int tid  = threadIdx.x;
    const int lane = tid & 31;
    const int wave = tid >> 5;          // 8 waves of 32
    const int mn   = lane & 15;         // M row (A/D) == N col (B/D) for this lane
    const int h    = lane >> 4;         // lane half

    const bool nvalid = (mn < CPC);
    const float bias1 = nvalid ? b1[g * CPC + mn] : 0.0f;
    const float bias2 = nvalid ? b2[g * CPC + mn] : 0.0f;

    // ---- stage x tile (f32 -> bf16) ----
    for (int i = tid; i < XS_SZ; i += 256) {
        int t = t0 + i;
        float v = (i < XNEED && t < T_IN) ? x[(b * CIN + g) * T_IN + t] : 0.0f;
        xs[i] = f2bf(v);
    }
    // ---- stage w1 slice as B[64 x 16] ----
    for (int i = tid; i < 16 * 64; i += 256) {
        int n = i >> 6, k = i & 63;
        float v = (n < CPC && k < K1) ? w1[(g * CPC + n) * K1 + k] : 0.0f;
        w1s[i] = f2bf(v);
    }
    // ---- stage w2 slice as B[512 x 16], K-permuted so each 32-K block sits in one ci row:
    //      blk 0..14 : ci = blk/3, koff = (blk%3)*32 + kl        (koff 0..95)
    //      blk 15    : kl<20 -> ci = kl>>2, koff = 96 + (kl&3)    (remainders), else zero
    for (int i = tid; i < 16 * 512; i += 256) {
        int n = i >> 9, j = i & 511;
        int blk = j >> 5, kl = j & 31;
        float v = 0.0f;
        if (n < CPC) {
            if (blk < 15) {
                int ci   = blk / 3;
                int koff = (blk - ci * 3) * 32 + kl;
                v = w2[(g * CPC + n) * KK2 + ci * K2 + koff];
            } else if (kl < 20) {
                int ci   = kl >> 2;
                int koff = 96 + (kl & 3);
                v = w2[(g * CPC + n) * KK2 + ci * K2 + koff];
            }
        }
        w2s[i] = f2bf(v);
    }
    for (int i = tid; i < YROW; i += 256) ys[5 * YROW + i] = 0; // zero pad row
    __syncthreads();

    // =================== stage 1: y = |conv_k50(x) + b1| ===================
    for (int s = wave; s < YTIMES / 16; s += 8) {
        v8f acc = {};
        #pragma unroll
        for (int kb = 0; kb < 2; ++kb) {
            ABReg A, Bm;
            const int abase = s * 16 + mn + 32 * kb;
            #pragma unroll
            for (int vr = 0; vr < 8; ++vr) {
                int kl = a_klocal(vr, h);
                unsigned int lo = xs[abase + kl];
                unsigned int hi = xs[abase + kl + 1];
                A.u[vr] = lo | (hi << 16);
            }
            const int bbase = mn * 64 + 32 * kb;
            #pragma unroll
            for (int vr = 0; vr < 8; ++vr)
                Bm.u[vr] = *(const unsigned int*)&w1s[bbase + a_klocal(vr, h)];
            acc = __builtin_amdgcn_wmma_f32_16x16x32_bf16(
                      false, A.v, false, Bm.v, (short)0, acc, false, false);
        }
        if (nvalid) {
            #pragma unroll
            for (int vr = 0; vr < 8; ++vr) {
                int tt = s * 16 + vr + 8 * h;
                ys[mn * YROW + tt] = f2bf(fabsf(acc[vr] + bias1));
            }
        }
    }
    __syncthreads();

    // =================== stage 2: z = conv_{5ci,k100}(y) + b2 ===================
    for (int s = wave; s < TILE_Z / 16; s += 8) {
        v8f acc = {};
        const int toff = 16 * s + mn;
        // blocks 0..14: contiguous per-lane A chunks (merge to ds_load_b128)
        #pragma unroll 3
        for (int kb = 0; kb < 15; ++kb) {
            int ci     = kb / 3;
            int kstart = (kb - ci * 3) * 32;
            const int abase = ci * YROW + kstart + toff;
            ABReg A, Bm;
            #pragma unroll
            for (int vr = 0; vr < 8; ++vr) {
                int kl = a_klocal(vr, h);
                unsigned int lo = ys[abase + kl];
                unsigned int hi = ys[abase + kl + 1];
                A.u[vr] = lo | (hi << 16);
            }
            const int bbase = mn * 512 + 32 * kb;
            #pragma unroll
            for (int vr = 0; vr < 8; ++vr)
                Bm.u[vr] = *(const unsigned int*)&w2s[bbase + a_klocal(vr, h)];
            acc = __builtin_amdgcn_wmma_f32_16x16x32_bf16(
                      false, A.v, false, Bm.v, (short)0, acc, false, false);
        }
        // block 15: remainders koff 96..99 per ci (pairs still contiguous per ci)
        {
            ABReg A, Bm;
            #pragma unroll
            for (int vr = 0; vr < 8; ++vr) {
                int kl = a_klocal(vr, h);
                int ci = kl >> 2;
                int e0 = (kl < 20) ? (ci * YROW + 96 + (kl & 3) + toff)
                                   : (5 * YROW + toff);
                unsigned int lo = ys[e0];
                unsigned int hi = ys[e0 + 1];
                A.u[vr] = lo | (hi << 16);
            }
            const int bbase = mn * 512 + 480;
            #pragma unroll
            for (int vr = 0; vr < 8; ++vr)
                Bm.u[vr] = *(const unsigned int*)&w2s[bbase + a_klocal(vr, h)];
            acc = __builtin_amdgcn_wmma_f32_16x16x32_bf16(
                      false, A.v, false, Bm.v, (short)0, acc, false, false);
        }
        if (nvalid) {
            #pragma unroll
            for (int vr = 0; vr < 8; ++vr) {
                int t = t0 + 16 * s + vr + 8 * h;
                if (t < T2)
                    out[((size_t)(b * OC + g * CPC + mn)) * T2 + t] = acc[vr] + bias2;
            }
        }
    }
}

extern "C" void kernel_launch(void* const* d_in, const int* in_sizes, int n_in,
                              void* d_out, int out_size, void* d_ws, size_t ws_size,
                              hipStream_t stream) {
    (void)in_sizes; (void)n_in; (void)d_ws; (void)ws_size; (void)out_size;
    const float* x  = (const float*)d_in[0];
    const float* w1 = (const float*)d_in[1];
    const float* b1 = (const float*)d_in[2];
    const float* w2 = (const float*)d_in[3];
    const float* b2 = (const float*)d_in[4];
    float* out = (float*)d_out;

    const int num_tiles = (T2 + TILE_Z - 1) / TILE_Z;     // 16
    dim3 grid(num_tiles, CIN, BATCH);                     // 16 x 64 x 32 workgroups
    dim3 block(256);                                      // 8 wave32
    envelope_fused_wmma<<<grid, block, 0, stream>>>(x, w1, b1, w2, b2, out);
}
